// KanAEClassifier_17858474016890
// MI455X (gfx1250) — matrix operand order repaired and checked by reading, not compiled, not run on hardware
//
#include <hip/hip_runtime.h>
#include <hip/hip_bf16.h>

typedef _Float16 v16h __attribute__((ext_vector_type(16)));
typedef _Float16 v8h  __attribute__((ext_vector_type(8)));
typedef _Float16 h2   __attribute__((ext_vector_type(2)));
typedef float    v8f  __attribute__((ext_vector_type(8)));

// ---------------- layer geometry (K includes silu + 8 spline coeffs per input, padded to 32) ----
// e1: In=78  Out=64  K=704 (pad 702->704) KT=22 NT=4
// e2: In=64  Out=32  K=576               KT=18 NT=2
// d1: In=32  Out=64  K=288               KT=9  NT=4
// d2: In=64  Out=78  K=576               KT=18 NT=5 (N pad 78->80)
#define OFF_E1 0
#define OFF_E2 45056          // 22*4*512
#define OFF_D1 63488          // +18*2*512
#define OFF_D2 81920          // +9*4*512
#define WS_HALVES 128000      // +18*5*512  (256000 bytes total in d_ws)

// ---------------- cubic B-spline bases, uniform knots t[j]=0.4j-2.2 -------------------------
__device__ __forceinline__ void bspline8(float x, float bs[8]) {
    float b[11];
#pragma unroll
    for (int j = 0; j < 11; ++j) {
        float t0 = 0.4f * j - 2.2f;
        b[j] = (x >= t0 && x < t0 + 0.4f) ? 1.0f : 0.0f;
    }
#pragma unroll
    for (int p = 1; p <= 3; ++p) {
        float inv = 1.0f / (0.4f * p);
#pragma unroll
        for (int j = 0; j < 11 - p; ++j) {
            float tj   = 0.4f * j - 2.2f;
            float tjp1 = tj + 0.4f * (p + 1);
            b[j] = (x - tj) * inv * b[j] + (tjp1 - x) * inv * b[j + 1];
        }
    }
#pragma unroll
    for (int c = 0; c < 8; ++c) bs[c] = b[c];
}

__device__ __forceinline__ float silu(float v) { return v / (1.0f + __expf(-v)); }

// write silu(v) + 8 bspline bases for input column `col` of row `row` into A staging (f16)
__device__ __forceinline__ void stage_elem(_Float16* Aw, int kpad, int In, int row, int col, float v) {
    _Float16* p = Aw + row * kpad;
    p[col] = (_Float16)silu(v);
    float bs[8];
    bspline8(v, bs);
    _Float16* q = p + In + col * 8;     // In even -> 4B aligned pair stores
#pragma unroll
    for (int c = 0; c < 4; ++c) {
        h2 t = { (_Float16)bs[2 * c], (_Float16)bs[2 * c + 1] };
        *(h2*)(q + 2 * c) = t;
    }
}

// wave-internal ordering fence: DS ops from one wave are processed in order; just stop the
// compiler from reordering and drain DScnt before cross-lane LDS reads.
__device__ __forceinline__ void wave_sync_lds() {
    __builtin_amdgcn_wave_barrier();
    asm volatile("s_wait_dscnt 0x0" ::: "memory");
    __builtin_amdgcn_wave_barrier();
}

// ---------------- WMMA GEMM over one wave's 16-row tile -------------------------------------
// A staged in LDS row-major f16 [16][kpad]; B pre-swizzled in d_ws: tile(kt,nt) = 512 halves,
// lane-major (lane*16 halves contiguous) matching v_wmma B fragment order.
template <int KT, int NT>
__device__ __forceinline__ void gemm_tile(const _Float16* Aw, int kpad,
                                          const _Float16* Bp, v8f (&acc)[NT], int lane) {
    const int koff = (lane & 16) ? 8 : 0;   // A: lanes 16-31 hold K 8-15 / 24-31
    const int arow = lane & 15;
    for (int kt = 0; kt < KT; ++kt) {
        const _Float16* ap = Aw + arow * kpad + kt * 32 + koff;
        v8h a0 = *(const v8h*)ap;            // K +0..7   (ds_load_b128)
        v8h a1 = *(const v8h*)(ap + 16);     // K +16..23 (ds_load_b128)
        v16h a = __builtin_shufflevector(a0, a1, 0,1,2,3,4,5,6,7,8,9,10,11,12,13,14,15);
        const _Float16* bp = Bp + (size_t)(kt * NT) * 512 + lane * 16;
#pragma unroll
        for (int nt = 0; nt < NT; ++nt) {
            v16h b = *(const v16h*)(bp + nt * 512);   // 2x global_load_b128
            acc[nt] = __builtin_amdgcn_wmma_f32_16x16x32_f16(
                false, a, false, b, (short)0, acc[nt], false, false);
        }
    }
}

// dump D fragments (f32) to LDS [16][ncols] so all lanes can restage them
template <int NT>
__device__ __forceinline__ void dump_act(const v8f (&acc)[NT], float* Fw, int ncols, int lane) {
    const int rbase = (lane & 16) ? 8 : 0;
    const int c0 = lane & 15;
#pragma unroll
    for (int nt = 0; nt < NT; ++nt)
#pragma unroll
        for (int v = 0; v < 8; ++v)
            Fw[(rbase + v) * ncols + nt * 16 + c0] = acc[nt][v];
}

// ---------------- weight packing: fold scaler, concat base|spline, f16, B-fragment swizzle ---
__global__ void pack_weights(const float* __restrict__ base_w, const float* __restrict__ spline_w,
                             const float* __restrict__ scaler, int In, int Out, int KT, int NT,
                             _Float16* __restrict__ dst) {
    int idx = blockIdx.x * blockDim.x + threadIdx.x;
    int total = KT * NT * 512;
    if (idx >= total) return;
    int tile = idx >> 9, rem = idx & 511;
    int lane = rem >> 4, t = rem & 15;
    int r = t >> 1, h = t & 1;
    int kt = tile / NT, nt = tile - kt * NT;
    int k = kt * 32 + ((lane & 16) ? 8 : 0) + ((t & 8) ? 16 : 0) + (t & 7);   // B: lanes 16-31 hold K 8-15 / 24-31
    int n = nt * 16 + (lane & 15);
    float val = 0.0f;
    if (n < Out) {
        if (k < In) {
            val = base_w[n * In + k];
        } else if (k < In * 9) {
            int i = (k - In) >> 3, c = (k - In) & 7;
            val = spline_w[(n * In + i) * 8 + c] * scaler[n * In + i];
        }
    }
    dst[idx] = (_Float16)val;
}

// ---------------- fused KAN autoencoder/classifier ------------------------------------------
#define WAVES 4
__global__ __launch_bounds__(128) void kan_fused(
    const float* __restrict__ x, const _Float16* __restrict__ wpack,
    const float* __restrict__ ln_w, const float* __restrict__ ln_b,
    const float* __restrict__ cls_w, const float* __restrict__ cls_b,
    float* __restrict__ out_logits, float* __restrict__ out_recon, float* __restrict__ out_z) {
    __shared__ _Float16 Ast[WAVES][16 * 704];   // per-wave A staging (max K=704)
    __shared__ float    Fst[WAVES][16 * 64];    // per-wave f32 activation staging (max 64 cols)

    const int tid = threadIdx.x;
    const int wave = tid >> 5, lane = tid & 31;
    _Float16* Aw = &Ast[wave][0];
    float*    Fw = &Fst[wave][0];
    const long rowbase = (long)blockIdx.x * (WAVES * 16) + wave * 16;

    // ---- stage e1 A from x: silu + bases, K padded 702->704 ----
    const float* xrow = x + rowbase * 78;
    for (int idx = lane; idx < 16 * 78; idx += 32) {     // contiguous: idx = row*78+col
        int row = idx / 78, col = idx - row * 78;
        stage_elem(Aw, 704, 78, row, col, xrow[idx]);
    }
    if (lane < 16) { Aw[lane * 704 + 702] = (_Float16)0.f; Aw[lane * 704 + 703] = (_Float16)0.f; }
    wave_sync_lds();

    // ---- e1: [16,704] x [704,64] ----
    v8f h[4] = {};
    gemm_tile<22, 4>(Aw, 704, wpack + OFF_E1, h, lane);
    dump_act<4>(h, Fw, 64, lane);
    wave_sync_lds();

    // ---- restage for e2 (In=64 -> K=576) ----
    for (int idx = lane; idx < 16 * 64; idx += 32) {
        int row = idx >> 6, col = idx & 63;
        stage_elem(Aw, 576, 64, row, col, Fw[idx]);
    }
    wave_sync_lds();

    // ---- e2: [16,576] x [576,32] ----
    v8f zr[2] = {};
    gemm_tile<18, 2>(Aw, 576, wpack + OFF_E2, zr, lane);
    dump_act<2>(zr, Fw, 32, lane);
    wave_sync_lds();

    // ---- LayerNorm(32) + logits + z store; lanes 0-15 each own a row ----
    if (lane < 16) {
        const int row = lane;
        float zv[32], s = 0.f;
#pragma unroll
        for (int j = 0; j < 32; ++j) { zv[j] = Fw[row * 32 + j]; s += zv[j]; }
        float mu = s * (1.0f / 32.0f), s2 = 0.f;
#pragma unroll
        for (int j = 0; j < 32; ++j) { float d = zv[j] - mu; s2 += d * d; }
        float rstd = rsqrtf(s2 * (1.0f / 32.0f) + 1e-5f);
        float l0 = cls_b[0], l1 = cls_b[1];
        const long grow = rowbase + row;
#pragma unroll
        for (int j = 0; j < 32; ++j) {
            float zn = (zv[j] - mu) * rstd * ln_w[j] + ln_b[j];
            Fw[row * 32 + j] = zn;
            out_z[grow * 32 + j] = zn;
            l0 += zn * cls_w[j];
            l1 += zn * cls_w[32 + j];
        }
        out_logits[grow * 2 + 0] = l0;
        out_logits[grow * 2 + 1] = l1;
    }
    wave_sync_lds();

    // ---- restage for d1 (In=32 -> K=288) ----
    for (int idx = lane; idx < 16 * 32; idx += 32) {
        int row = idx >> 5, col = idx & 31;
        stage_elem(Aw, 288, 32, row, col, Fw[idx]);
    }
    wave_sync_lds();

    // ---- d1: [16,288] x [288,64] ----
    v8f hd[4] = {};
    gemm_tile<9, 4>(Aw, 288, wpack + OFF_D1, hd, lane);
    dump_act<4>(hd, Fw, 64, lane);
    wave_sync_lds();

    // ---- restage for d2 (In=64 -> K=576) ----
    for (int idx = lane; idx < 16 * 64; idx += 32) {
        int row = idx >> 6, col = idx & 63;
        stage_elem(Aw, 576, 64, row, col, Fw[idx]);
    }
    wave_sync_lds();

    // ---- d2: [16,576] x [576,80] (cols 78,79 are zero-weight pad) ----
    v8f rc[5] = {};
    gemm_tile<18, 5>(Aw, 576, wpack + OFF_D2, rc, lane);

    const int rbase = (lane & 16) ? 8 : 0;
    const int c0 = lane & 15;
#pragma unroll
    for (int nt = 0; nt < 5; ++nt) {
        int col = nt * 16 + c0;
        if (col < 78) {
#pragma unroll
            for (int v = 0; v < 8; ++v)
                out_recon[(rowbase + rbase + v) * 78 + col] = rc[nt][v];
        }
    }
}

extern "C" void kernel_launch(void* const* d_in, const int* in_sizes, int n_in,
                              void* d_out, int out_size, void* d_ws, size_t ws_size,
                              hipStream_t stream) {
    const float* x           = (const float*)d_in[0];
    const float* e1_base_w   = (const float*)d_in[1];
    const float* e1_spline_w = (const float*)d_in[2];
    const float* e1_scaler   = (const float*)d_in[3];
    const float* e2_base_w   = (const float*)d_in[4];
    const float* e2_spline_w = (const float*)d_in[5];
    const float* e2_scaler   = (const float*)d_in[6];
    const float* ln_w        = (const float*)d_in[7];
    const float* ln_b        = (const float*)d_in[8];
    const float* cls_w       = (const float*)d_in[9];
    const float* cls_b       = (const float*)d_in[10];
    const float* d1_base_w   = (const float*)d_in[11];
    const float* d1_spline_w = (const float*)d_in[12];
    const float* d1_scaler   = (const float*)d_in[13];
    const float* d2_base_w   = (const float*)d_in[14];
    const float* d2_spline_w = (const float*)d_in[15];
    const float* d2_scaler   = (const float*)d_in[16];

    const int Bn = in_sizes[0] / 78;
    _Float16* ws = (_Float16*)d_ws;
    float* out = (float*)d_out;
    float* out_logits = out;
    float* out_recon  = out + (size_t)Bn * 2;
    float* out_z      = out + (size_t)Bn * 80;

    // pack each layer's combined weights into pre-swizzled f16 B-fragments in d_ws
    pack_weights<<<(22 * 4 * 512 + 255) / 256, 256, 0, stream>>>(e1_base_w, e1_spline_w, e1_scaler, 78, 64, 22, 4, ws + OFF_E1);
    pack_weights<<<(18 * 2 * 512 + 255) / 256, 256, 0, stream>>>(e2_base_w, e2_spline_w, e2_scaler, 64, 32, 18, 2, ws + OFF_E2);
    pack_weights<<<( 9 * 4 * 512 + 255) / 256, 256, 0, stream>>>(d1_base_w, d1_spline_w, d1_scaler, 32, 64,  9, 4, ws + OFF_D1);
    pack_weights<<<(18 * 5 * 512 + 255) / 256, 256, 0, stream>>>(d2_base_w, d2_spline_w, d2_scaler, 64, 78, 18, 5, ws + OFF_D2);

    // fused network: 64 rows per block (4 waves x 16 rows), B=131072 -> 2048 blocks
    kan_fused<<<Bn / 64, 128, 0, stream>>>(x, ws, ln_w, ln_b, cls_w, cls_b,
                                           out_logits, out_recon, out_z);
}
